// LSTM_71236327572079
// MI455X (gfx1250) — matrix operand order repaired and checked by reading, not compiled, not run on hardware
//
#include <hip/hip_runtime.h>
#include <hip/hip_bf16.h>

typedef __attribute__((ext_vector_type(16))) _Float16 v16h;
typedef __attribute__((ext_vector_type(8)))  _Float16 h8;
typedef __attribute__((ext_vector_type(4)))  _Float16 h4;
typedef __attribute__((ext_vector_type(8)))  float    v8f;
typedef __attribute__((ext_vector_type(4)))  float    f4;

namespace {
constexpr int NB   = 8;     // batch
constexpr int NT   = 20;    // timesteps
constexpr int NN   = 1000;  // stocks
constexpr int DIN  = 64;
constexpr int DD   = 64;
constexpr int KK   = 128;   // DIN + DD
constexpr int KWP  = 136;   // padded K stride for W in LDS (f16 units, 272B, 16B aligned)
constexpr int KXP  = 136;   // padded K stride for xh in LDS
}

static __device__ __forceinline__ v16h combine16(h8 lo, h8 hi) {
  v16h r;
#pragma unroll
  for (int i = 0; i < 8; ++i) { r[i] = lo[i]; r[8 + i] = hi[i]; }
  return r;
}

// Native v_tanh_f32 on gfx1250 if the builtin exists; branch-free fallback.
static __device__ __forceinline__ float fast_tanh(float x) {
#if __has_builtin(__builtin_amdgcn_tanhf)
  return __builtin_amdgcn_tanhf(x);
#else
  float e = __expf(2.0f * x);
  return 1.0f - 2.0f / (e + 1.0f);
#endif
}

static __device__ __forceinline__ float fast_sigmoid(float x) {
  return 0.5f * fast_tanh(0.5f * x) + 0.5f;
}

__global__ __launch_bounds__(128)
void lstm_att_kernel(const float* __restrict__ x,
                     const float* __restrict__ Wi, const float* __restrict__ bi,
                     const float* __restrict__ Wo, const float* __restrict__ bo,
                     const float* __restrict__ Wf, const float* __restrict__ bf,
                     const float* __restrict__ Wc, const float* __restrict__ bc,
                     const float* __restrict__ Wt,
                     float* __restrict__ out)
{
  // one block per stock; 4 waves; wave w owns output columns [16w, 16w+16)
  __shared__ __align__(16) _Float16 Wbuf[DD * KWP];        // 17408 B (one gate, transposed [col][k])
  __shared__ __align__(16) _Float16 xh[16 * KXP];          //  4352 B ([row(M)][k]), rows 8..15 = 0 pad
  __shared__ __align__(16) float    hist[NT * NB * DD];    // 40960 B  h history for attention
  __shared__ __align__(16) float    sc[NB * 32];           //  1024 B  attention scores

  const int n    = blockIdx.x;
  const int tid  = threadIdx.x;
  const int lane = tid & 31;
  const int wave = tid >> 5;
  const int half = lane >> 4;         // 0: lanes 0-15, 1: lanes 16-31
  const int lr   = lane & 15;
  const int col  = wave * 16 + lr;    // d-column this lane owns in B/C/D fragments

  const float* Wg[4] = {Wi, Wo, Wf, Wc};
  const float* bg[4] = {bi, bo, bf, bc};

  // zero xh once: covers h-part at t=0 and padding rows 8..15 forever
  for (int i = tid; i < 16 * KXP; i += 128) xh[i] = (_Float16)0.0f;

  // per-lane biases (depend only on column)
  float bias[4];
#pragma unroll
  for (int g = 0; g < 4; ++g) bias[g] = bg[g][n * DD + col];

  // ---- stage weights gate-by-gate, hoist B fragments into registers ----
  // bfrag: 4 gates x 4 K-steps x v16h = 128 VGPRs, live across whole t-loop
  v16h bfrag[4][4];
#pragma unroll
  for (int g = 0; g < 4; ++g) {
    __syncthreads();
    const float* W = Wg[g] + (size_t)n * KK * DD;   // [k][col], col contiguous
    for (int i = tid; i < KK * DD; i += 128) {
      int k = i >> 6, c = i & 63;                    // coalesced global read
      Wbuf[c * KWP + k] = (_Float16)W[i];            // transposed LDS write
    }
    __syncthreads();
#pragma unroll
    for (int ks = 0; ks < 4; ++ks) {
      // B 32x16 f16, contiguous k order (matches A): lane = col, lanes0-15 K=[32ks,32ks+16), lanes16-31 K+16
      int koff = ks * 32 + half * 16;
      const _Float16* p = &Wbuf[col * KWP + koff];
      h8 lo = *(const h8*)p;
      h8 hi = *(const h8*)(p + 8);
      bfrag[g][ks] = combine16(lo, hi);
    }
  }
  __syncthreads();

  // c state: C/D layout — lane(0-15) VGPR r holds (batch=r, d=col); lanes 16-31 are pad rows
  v8f cstate = {};

  // per-thread x staging coordinates: one float4 load + one b64 LDS store per t
  const int xb = tid >> 4;            // batch row 0..7
  const int xd = (tid & 15) << 2;     // d offset 0,4,...,60

  for (int t = 0; t < NT; ++t) {
    // write x_t into xh rows 0..7 (cols 0..63), coalesced 16B global loads
    {
      f4 xv = *(const f4*)&x[(((size_t)xb * NT + t) * NN + n) * DIN + xd];
      h4 hv;
#pragma unroll
      for (int i = 0; i < 4; ++i) hv[i] = (_Float16)xv[i];
      *(h4*)&xh[xb * KXP + xd] = hv;
    }
    __syncthreads();   // xh (x-part this t, h-part from t-1) ready

    // prefetch all A fragments (8 x ds_load_b128), then run 16 WMMAs back-to-back
    v16h afrag[4];
#pragma unroll
    for (int ks = 0; ks < 4; ++ks) {
      // A 16x32 f16, contiguous k order (matches B): lane row = lr; 16 contiguous K at 32ks+16*half
      int koff = ks * 32 + half * 16;
      const _Float16* p = &xh[lr * KXP + koff];
      h8 a0 = *(const h8*)p;
      h8 a1 = *(const h8*)(p + 8);
      afrag[ks] = combine16(a0, a1);
    }

    v8f acc[4] = {};
#pragma unroll
    for (int ks = 0; ks < 4; ++ks) {
#pragma unroll
      for (int g = 0; g < 4; ++g)
        acc[g] = __builtin_amdgcn_wmma_f32_16x16x32_f16(
            false, afrag[ks], false, bfrag[g][ks], (short)0, acc[g], false, false);
    }

    // elementwise LSTM cell, fully in registers, branch-free trans ops
    float h2[8];
#pragma unroll
    for (int r = 0; r < 8; ++r) {
      float it = fast_sigmoid(acc[0][r] + bias[0]);
      float ot = fast_sigmoid(acc[1][r] + bias[1]);
      float ft = fast_sigmoid(acc[2][r] + bias[2]);
      float ch = fast_tanh(acc[3][r] + bias[3]);
      float c2 = ft * cstate[r] + it * ch;
      cstate[r] = c2;
      h2[r] = ot * fast_tanh(c2);
    }

    __syncthreads();   // all waves done reading xh before h-part overwrite
    if (lane < 16) {   // only lanes 0-15 hold real batch rows
#pragma unroll
      for (int r = 0; r < 8; ++r) {
        xh[r * KXP + DIN + col]       = (_Float16)h2[r];
        hist[(t * NB + r) * DD + col] = h2[r];
      }
    }
    // next iteration's top-of-loop __syncthreads() publishes these writes
  }
  __syncthreads();

  // ---- temporal attention: scores[b,t] = <h[b,t,:], Wt[n,t,:]> ----
  for (int s = tid; s < NB * NT; s += 128) {
    int b = s / NT, t = s - b * NT;
    const float* wt = Wt + ((size_t)n * NT + t) * DD;
    float a = 0.0f;
    for (int d = 0; d < DD; ++d) a += hist[(t * NB + b) * DD + d] * wt[d];
    sc[b * 32 + t] = a;
  }
  __syncthreads();

  if (tid < NB) {   // softmax over t, one thread per batch row
    int b = tid;
    float m = -3.0e38f;
    for (int t = 0; t < NT; ++t) m = fmaxf(m, sc[b * 32 + t]);
    float ssum = 0.0f;
    for (int t = 0; t < NT; ++t) {
      float e = __expf(sc[b * 32 + t] - m);
      sc[b * 32 + t] = e;
      ssum += e;
    }
    float inv = 1.0f / ssum;
    for (int t = 0; t < NT; ++t) sc[b * 32 + t] *= inv;
  }
  __syncthreads();

  // out[b,n,d] = sum_t att[b,t] * h[b,t,d]   (coalesced in d)
#pragma unroll
  for (int rep = 0; rep < 4; ++rep) {
    int idx = tid + rep * 128;               // 0..511
    int b = idx >> 6, d = idx & 63;
    float a = 0.0f;
    for (int t = 0; t < NT; ++t) a += sc[b * 32 + t] * hist[(t * NB + b) * DD + d];
    out[((size_t)b * NN + n) * DD + d] = a;
  }
}

extern "C" void kernel_launch(void* const* d_in, const int* in_sizes, int n_in,
                              void* d_out, int out_size, void* d_ws, size_t ws_size,
                              hipStream_t stream) {
  (void)in_sizes; (void)n_in; (void)d_ws; (void)ws_size; (void)out_size;
  const float* x  = (const float*)d_in[0];
  const float* Wi = (const float*)d_in[1];
  const float* bi = (const float*)d_in[2];
  const float* Wo = (const float*)d_in[3];
  const float* bo = (const float*)d_in[4];
  const float* Wf = (const float*)d_in[5];
  const float* bf = (const float*)d_in[6];
  const float* Wc = (const float*)d_in[7];
  const float* bc = (const float*)d_in[8];
  const float* Wt = (const float*)d_in[9];
  float* out = (float*)d_out;

  lstm_att_kernel<<<dim3(NN), dim3(128), 0, stream>>>(
      x, Wi, bi, Wo, bo, Wf, bf, Wc, bc, Wt, out);
}
